// DSAG_58025008169585
// MI455X (gfx1250) — matrix operand order repaired and checked by reading, not compiled, not run on hardware
//
#include <hip/hip_runtime.h>
#include <hip/hip_bf16.h>
#include <math.h>

// ---------------------------------------------------------------------------
// DSAG forward for MI455X (gfx1250, wave32).
// Heavy GEMMs on v_wmma_f32_16x16x32_bf16 (bf16 A/B, f32 accum).
// Compute-bound (~120 GFLOP vs ~230MB of HBM traffic after folding the exo
// affordance conv into a weighted mean and the ego modulation into the GEMM
// epilogue), so WMMA issue rate is the limiter.
// GEMM codegen: compile-time strides (immediate load offsets / b128 loads on
// unit-stride K operands), clamped OOB rows/cols (no exec divergence),
// batched pipelined loads, 2 wmma per K-step sharing the A fetch.
// ---------------------------------------------------------------------------

typedef __attribute__((ext_vector_type(16))) __bf16 v16bf;
typedef __attribute__((ext_vector_type(8)))  float  v8f;

// ----------------------------- reductions ----------------------------------
__device__ __forceinline__ float wave_sum(float v) {
#pragma unroll
  for (int off = 16; off > 0; off >>= 1) v += __shfl_xor(v, off, 32);
  return v;
}
__device__ __forceinline__ float wave_max(float v) {
#pragma unroll
  for (int off = 16; off > 0; off >>= 1) v = fmaxf(v, __shfl_xor(v, off, 32));
  return v;
}
__device__ __forceinline__ float block_sum(float v, float* sred /*>=33*/) {
  __syncthreads();
  int lane = threadIdx.x & 31, w = threadIdx.x >> 5;
  v = wave_sum(v);
  if (lane == 0) sred[w] = v;
  __syncthreads();
  int nw = (blockDim.x + 31) >> 5;
  float t = (threadIdx.x < (unsigned)nw) ? sred[threadIdx.x] : 0.f;
  if (w == 0) t = wave_sum(t);
  if (threadIdx.x == 0) sred[32] = t;
  __syncthreads();
  return sred[32];
}

// --------------------- per-lane tile fragment loaders ----------------------
// A fragment: f[0..15] = p[0..15]  (stride in K)
template <int STRIDE>
__device__ __forceinline__ void load16_a(const float* __restrict__ p, float* f) {
  if constexpr (STRIDE == 1) {
    const float4* q = (const float4*)p;          // 16B aligned at all call sites
    float4 v0 = q[0], v1 = q[1], v2 = q[2], v3 = q[3];
    f[0] = v0.x;  f[1] = v0.y;  f[2] = v0.z;  f[3] = v0.w;
    f[4] = v1.x;  f[5] = v1.y;  f[6] = v1.z;  f[7] = v1.w;
    f[8] = v2.x;  f[9] = v2.y;  f[10] = v2.z; f[11] = v2.w;
    f[12] = v3.x; f[13] = v3.y; f[14] = v3.z; f[15] = v3.w;
  } else {
#pragma unroll
    for (int i = 0; i < 8; ++i) {                // immediate offsets
      f[i]     = p[i * STRIDE];
      f[i + 8] = p[(8 + i) * STRIDE];
    }
  }
}
// B fragment: f[0..15] = p[0..15] (stride in K)
template <int STRIDE>
__device__ __forceinline__ void load16_b(const float* __restrict__ p, float* f) {
  if constexpr (STRIDE == 1) {
    const float4* q = (const float4*)p;
    float4 v0 = q[0], v1 = q[1], v2 = q[2], v3 = q[3];
    f[0] = v0.x;  f[1] = v0.y;  f[2] = v0.z;  f[3] = v0.w;
    f[4] = v1.x;  f[5] = v1.y;  f[6] = v1.z;  f[7] = v1.w;
    f[8] = v2.x;  f[9] = v2.y;  f[10] = v2.z; f[11] = v2.w;
    f[12] = v3.x; f[13] = v3.y; f[14] = v3.z; f[15] = v3.w;
  } else {
#pragma unroll
    for (int i = 0; i < 16; ++i) f[i] = p[i * STRIDE];
  }
}

// ----------------------- generic batched WMMA GEMM -------------------------
// C[b,m,n] = alpha * exp?(scale) * colscale[b,n] * sum_k A[b,m,k]*B[b,k,n]
//            + bias_m[m] + bias_n[n]
// One wave computes a 16x32 output tile (two 16x16 WMMA tiles sharing A).
template <int SAM, int SAK, int SBK, int SBN>
__global__ void __launch_bounds__(32)
dsag_gemm(const float* __restrict__ A, long long sAb,
          const float* __restrict__ B, long long sBb,
          float* __restrict__ C, int sCm, int sCn, long long sCb,
          const float* __restrict__ bias_m, const float* __restrict__ bias_n,
          const float* __restrict__ colscale,
          const float* __restrict__ scale_ptr, float alpha,
          int M, int N, int K) {
  const int lane = threadIdx.x & 31;
  const int m0 = blockIdx.x * 16;
  const int n0 = blockIdx.y * 32;
  const int b  = blockIdx.z;

  const int l15 = lane & 15;
  const int hi  = lane >> 4;
  const int khA = hi ? 16 : 0;   // A: lanes16-31 carry K 16..31
  const int kbB = hi ? 16 : 0;   // B: lanes16-31 carry K 16..31

  // Clamp: OOB rows/cols compute garbage that is never stored.
  const int mi  = min(m0 + l15, M - 1);
  const int ni0 = min(n0 + l15, N - 1);
  const int ni1 = min(n0 + 16 + l15, N - 1);

  const float* arow = A + (long long)b * sAb + (long long)mi  * SAM + (long long)khA * SAK;
  const float* bc0  = B + (long long)b * sBb + (long long)ni0 * SBN + (long long)kbB * SBK;
  const float* bc1  = B + (long long)b * sBb + (long long)ni1 * SBN + (long long)kbB * SBK;

  v8f acc0 = {0.f, 0.f, 0.f, 0.f, 0.f, 0.f, 0.f, 0.f};
  v8f acc1 = {0.f, 0.f, 0.f, 0.f, 0.f, 0.f, 0.f, 0.f};

  const int K32 = K & ~31;
  for (int kk = 0; kk < K32; kk += 32) {
    float af[16], f0[16], f1[16];
    load16_a<SAK>(arow, af);
    load16_b<SBK>(bc0, f0);
    load16_b<SBK>(bc1, f1);
    v16bf av, bv0, bv1;
#pragma unroll
    for (int i = 0; i < 16; ++i) {
      av[i]  = (__bf16)af[i];
      bv0[i] = (__bf16)f0[i];
      bv1[i] = (__bf16)f1[i];
    }
    acc0 = __builtin_amdgcn_wmma_f32_16x16x32_bf16(false, av, false, bv0,
                                                   (short)0, acc0, false, false);
    acc1 = __builtin_amdgcn_wmma_f32_16x16x32_bf16(false, av, false, bv1,
                                                   (short)0, acc1, false, false);
    arow += 32 * SAK;
    bc0  += 32 * SBK;
    bc1  += 32 * SBK;
  }

  if (K32 < K) {  // single tail iteration: clamped loads + branchless zeroing
    const int kk = K32;
    float af[16], f0[16], f1[16];
#pragma unroll
    for (int i = 0; i < 8; ++i) {
      int k1 = kk + khA + i, k2 = k1 + 8;
      float v1 = arow[(min(k1, K - 1) - kk - khA) * SAK];
      float v2 = arow[(min(k2, K - 1) - kk - khA) * SAK];
      af[i]     = (k1 < K) ? v1 : 0.f;
      af[i + 8] = (k2 < K) ? v2 : 0.f;
    }
#pragma unroll
    for (int i = 0; i < 16; ++i) {
      int k = kk + kbB + i;
      int o = (min(k, K - 1) - kk - kbB) * SBK;
      float v0 = bc0[o], v1 = bc1[o];
      f0[i] = (k < K) ? v0 : 0.f;
      f1[i] = (k < K) ? v1 : 0.f;
    }
    v16bf av, bv0, bv1;
#pragma unroll
    for (int i = 0; i < 16; ++i) {
      av[i]  = (__bf16)af[i];
      bv0[i] = (__bf16)f0[i];
      bv1[i] = (__bf16)f1[i];
    }
    acc0 = __builtin_amdgcn_wmma_f32_16x16x32_bf16(false, av, false, bv0,
                                                   (short)0, acc0, false, false);
    acc1 = __builtin_amdgcn_wmma_f32_16x16x32_bf16(false, av, false, bv1,
                                                   (short)0, acc1, false, false);
  }

  float s = alpha;
  if (scale_ptr) s *= __expf(*scale_ptr);
  const int mb = m0 + (hi ? 8 : 0);

#pragma unroll
  for (int t = 0; t < 2; ++t) {
    const int n = n0 + t * 16 + l15;
    if (n < N) {
      float cs = colscale ? colscale[(long long)b * N + n] : 1.f;
      float bn = bias_n ? bias_n[n] : 0.f;
      float* cptr = C + (long long)b * sCb + (long long)n * sCn;
      const v8f& acc = t ? acc1 : acc0;
#pragma unroll
      for (int r = 0; r < 8; ++r) {
        int m = mb + r;
        if (m < M) {
          float bm = bias_m ? bias_m[m] : 0.f;
          cptr[(long long)m * sCm] = s * cs * acc[r] + bm + bn;
        }
      }
    }
  }
}

// ------------------------------ small kernels ------------------------------
__global__ void dsag_l2rows(const float* __restrict__ src, float* __restrict__ dst, int L) {
  __shared__ float sred[33];
  const float* p = src + (long long)blockIdx.x * L;
  float sq = 0.f;
  for (int e = threadIdx.x; e < L; e += blockDim.x) { float v = p[e]; sq += v * v; }
  float tot = block_sum(sq, sred);
  float inv = 1.f / fmaxf(sqrtf(tot), 1e-12f);
  float* q = dst + (long long)blockIdx.x * L;
  for (int e = threadIdx.x; e < L; e += blockDim.x) q[e] = p[e] * inv;
}

// img[b,:] = l2norm( mean_s tok[b,s,:] )   tok layout [B,S,E]
__global__ void dsag_img_pool_norm(const float* __restrict__ tok, float* __restrict__ imgn,
                                   int S, int E) {
  extern __shared__ float sh[];            // E vals + 33 reduce
  float* vals = sh;
  float* sred = sh + E;
  int b = blockIdx.x;
  const float* t = tok + (long long)b * S * E;
  float sq = 0.f;
  for (int e = threadIdx.x; e < E; e += blockDim.x) {
    float s = 0.f;
    for (int ss = 0; ss < S; ++ss) s += t[(long long)ss * E + e];
    s /= (float)S;
    vals[e] = s;
    sq += s * s;
  }
  float tot = block_sum(sq, sred);
  float inv = 1.f / fmaxf(sqrtf(tot), 1e-12f);
  for (int e = threadIdx.x; e < E; e += blockDim.x)
    imgn[(long long)b * E + e] = vals[e] * inv;
}

// att[b,s] = sigmoid( l2norm_s( ls * dot(l2norm_e(tok[b,s,:]), txtn[label[b%32],:]) ) )
__global__ void dsag_att(const float* __restrict__ tok, const float* __restrict__ txtn,
                         const int* __restrict__ label, const float* __restrict__ ls_ptr,
                         float* __restrict__ att, float* __restrict__ cs,
                         int S, int E) {
  __shared__ float r[196];
  __shared__ float sred[33];
  const int b = blockIdx.x;
  const int lane = threadIdx.x & 31, w = threadIdx.x >> 5, nw = blockDim.x >> 5;
  const int lab = label[b & 31];
  const float* t = txtn + (long long)lab * E;
  const float* tb = tok + (long long)b * S * E;
  for (int s = w; s < S; s += nw) {
    const float* p = tb + (long long)s * E;
    float dot = 0.f, sq = 0.f;
    for (int e = lane; e < E; e += 32) { float v = p[e]; dot += v * t[e]; sq += v * v; }
    dot = wave_sum(dot);
    sq  = wave_sum(sq);
    if (lane == 0) r[s] = dot / fmaxf(sqrtf(sq), 1e-12f);
  }
  float ls = __expf(*ls_ptr);
  float part = 0.f;
  __syncthreads();
  for (int s = threadIdx.x; s < S; s += blockDim.x) { float q = ls * r[s]; part += q * q; }
  float nrm = fmaxf(sqrtf(block_sum(part, sred)), 1e-12f);
  for (int s = threadIdx.x; s < S; s += blockDim.x) {
    float q = ls * r[s] / nrm;
    float a = 1.f / (1.f + __expf(-q));
    att[(long long)b * S + s] = a;
    if (cs) cs[(long long)b * S + s] = 0.5f * a + 0.5f;
  }
}

// wm[b,c] = (1/S) * sum_s (0.5*att[b,s]+0.5) * x[b,c,s]
__global__ void dsag_wmean(const float* __restrict__ x, const float* __restrict__ att,
                           float* __restrict__ wm, long long total, int C, int S) {
  long long idx = (long long)blockIdx.x * blockDim.x + threadIdx.x;
  if (idx >= total) return;
  int b = (int)(idx / C), c = (int)(idx % C);
  const float* xp = x + ((long long)b * C + c) * S;
  const float* ap = att + (long long)b * S;
  float acc = 0.f;
  for (int s = 0; s < S; ++s) acc += (0.5f * ap[s] + 0.5f) * xp[s];
  wm[idx] = acc / (float)S;
}

// mean over last dim S of [rows, S]
__global__ void dsag_mean_last(const float* __restrict__ src, float* __restrict__ dst,
                               long long rows, int S) {
  long long idx = (long long)blockIdx.x * blockDim.x + threadIdx.x;
  if (idx >= rows) return;
  const float* p = src + idx * S;
  float acc = 0.f;
  for (int s = 0; s < S; ++s) acc += p[s];
  dst[idx] = acc / (float)S;
}

// cam[b,s] = (1/O) * sum_o fc_w[label[b],o] * ebr[b,o,s]
__global__ void dsag_cam(const float* __restrict__ ebr, const float* __restrict__ fc_w,
                         const int* __restrict__ label, float* __restrict__ cam,
                         int B, int O, int S) {
  int idx = blockIdx.x * blockDim.x + threadIdx.x;
  if (idx >= B * S) return;
  int b = idx / S, s = idx % S;
  const float* wrow = fc_w + (long long)label[b] * O;
  const float* p = ebr + (long long)b * O * S + s;
  float acc = 0.f;
  for (int o = 0; o < O; ++o) acc += wrow[o] * p[(long long)o * S];
  cam[idx] = acc / (float)O;
}

// softmax in-place over rows of length S; one wave per row
__global__ void dsag_softmax_rows(float* __restrict__ mu, long long rows, int S) {
  long long row = (long long)blockIdx.x * (blockDim.x >> 5) + (threadIdx.x >> 5);
  int lane = threadIdx.x & 31;
  if (row >= rows) return;
  float* p = mu + row * S;
  float m = -3.4e38f;
  for (int t = lane; t < S; t += 32) m = fmaxf(m, p[t]);
  m = wave_max(m);
  float sum = 0.f;
  for (int t = lane; t < S; t += 32) { float e = __expf(p[t] - m); p[t] = e; sum += e; }
  sum = wave_sum(sum);
  float inv = 1.f / sum;
  for (int t = lane; t < S; t += 32) p[t] *= inv;
}

__global__ void dsag_mu_colnorm(float* __restrict__ mu, int B, int S) {
  int idx = blockIdx.x * blockDim.x + threadIdx.x;
  if (idx >= B * S) return;
  int b = idx / S, t = idx % S;
  float* p = mu + (long long)b * S * S + t;
  float s = 0.f;
  for (int i = 0; i < S; ++i) s += p[(long long)i * S];
  float inv = 1.f / s;
  for (int i = 0; i < S; ++i) p[(long long)i * S] *= inv;
}

__global__ void dsag_mu_rownorm(float* __restrict__ mu, int B, int S) {
  int idx = blockIdx.x * blockDim.x + threadIdx.x;
  if (idx >= B * S) return;
  float* p = mu + (long long)idx * S;
  float s = 0.f;
  for (int i = 0; i < S; ++i) s += p[i];
  float inv = 1.f / s;
  for (int i = 0; i < S; ++i) p[i] *= inv;
}

__global__ void dsag_mu_sym(const float* __restrict__ a, float* __restrict__ o,
                            long long total, int S) {
  long long idx = (long long)blockIdx.x * blockDim.x + threadIdx.x;
  if (idx >= total) return;
  long long bss = (long long)S * S;
  long long b = idx / bss, r = idx % bss;
  int s = (int)(r / S), t = (int)(r % S);
  o[idx] = 0.5f * (a[idx] + a[b * bss + (long long)t * S + s]);
}

__global__ void dsag_transpose(const float* __restrict__ src, float* __restrict__ dst,
                               int M, int N) {
  int idx = blockIdx.x * blockDim.x + threadIdx.x;
  if (idx >= M * N) return;
  int m = idx / N, n = idx % N;
  dst[(long long)n * M + m] = src[idx];
}

// sim_loss[b] = relu(1 - cos(attego[b%32,:], attexo[b,:]))  ; one wave per b
__global__ void dsag_sim_loss(const float* __restrict__ attg, const float* __restrict__ attx,
                              float* __restrict__ sim, int S) {
  int b = blockIdx.x, lane = threadIdx.x;
  const float* a = attg + (long long)(b & 31) * S;
  const float* e = attx + (long long)b * S;
  float dot = 0.f, na = 0.f, ne = 0.f;
  for (int s = lane; s < S; s += 32) {
    float av = a[s], ev = e[s];
    dot += av * ev; na += av * av; ne += ev * ev;
  }
  dot = wave_sum(dot); na = wave_sum(na); ne = wave_sum(ne);
  if (lane == 0) {
    float c = dot / fmaxf(sqrtf(na) * sqrtf(ne), 1e-8f);
    sim[b] = fmaxf(1.f - c, 0.f);
  }
}

// ------------------------------- launcher ----------------------------------
extern "C" void kernel_launch(void* const* d_in, const int* in_sizes, int n_in,
                              void* d_out, int out_size, void* d_ws, size_t ws_size,
                              hipStream_t stream) {
  (void)in_sizes; (void)n_in; (void)out_size; (void)ws_size;

  const float* exo_proj = (const float*)d_in[0];   // [96,2048,14,14]
  const float* ego_proj = (const float*)d_in[1];   // [32,2048,14,14]
  const float* text     = (const float*)d_in[2];   // [36,1024]
  const float* Wp_exo   = (const float*)d_in[3];   // [2048,1024]
  const float* Wp_ego   = (const float*)d_in[4];   // [2048,1024]
  const float* Waff_exo = (const float*)d_in[5];   // [512,2048]
  const float* baff_exo = (const float*)d_in[6];   // [512]
  const float* Waff_ego = (const float*)d_in[7];   // [512,2048]
  const float* baff_ego = (const float*)d_in[8];   // [512]
  const float* fc_w     = (const float*)d_in[9];   // [36,512]
  const float* fc_b     = (const float*)d_in[10];  // [36]
  const float* fce_w    = (const float*)d_in[11];  // [36,512]
  const float* fce_b    = (const float*)d_in[12];  // [36]
  const float* ls_ego   = (const float*)d_in[13];  // scalar (log)
  const float* ls_exo   = (const float*)d_in[14];  // scalar (log)
  const int*   label    = (const int*)d_in[15];    // [32]

  constexpr int EB = 32, C = 2048, S = 196, BEXO = 96, E = 1024, NC = 36, C2 = 512;

  float* out = (float*)d_out;
  float* o_exo_score = out;                        // [96,36]
  float* o_ego_score = out + 3456;                 // [32,36]
  float* o_lpt       = out + 4608;                 // [36,32]
  float* o_lpi       = out + 5760;                 // [32,36]
  float* o_lpt_exo   = out + 6912;                 // [36,96]
  float* o_lpi_exo   = out + 10368;                // [96,36]
  float* o_sim       = out + 13824;                // [96]
  float* o_cam       = out + 13920;                // [32,14,14]
  float* o_ebr       = out + 20192;                // [32,512,196]
  float* o_mu        = out + 3231456;              // [32,196,196]

  float* w = (float*)d_ws;
  float* txtn = w;  w += (size_t)NC * E;
  float* tokx = w;  w += (size_t)BEXO * S * E;     // [96,196,1024]
  float* tokg = w;  w += (size_t)EB * S * E;       // [32,196,1024]
  float* imgx = w;  w += (size_t)BEXO * E;
  float* imgg = w;  w += (size_t)EB * E;
  float* attx = w;  w += (size_t)BEXO * S;
  float* attg = w;  w += (size_t)EB * S;
  float* csg  = w;  w += (size_t)EB * S;           // 0.5*attego+0.5
  float* wmx  = w;  w += (size_t)BEXO * C;         // exo weighted mean [96,2048]
  float* affx = w;  w += (size_t)BEXO * C2;        // [96,512]
  float* egm  = w;  w += (size_t)EB * C2;          // [32,512]
  float* muA  = w;  w += (size_t)EB * S * S;
  float* muB  = w;  w += (size_t)EB * S * S;

  const dim3 b32(32), b256(256);
  const float* nul = nullptr;

  // 1) normalize text features
  dsag_l2rows<<<NC, b256, 0, stream>>>(text, txtn, E);

  // 2/3) tok[b,s,e] = sum_c x[b,c,s] * Wp[c,e]   A:(sAm=1,sAk=S) B:(sBk=E,sBn=1)
  dsag_gemm<1, S, E, 1><<<dim3(13, 32, BEXO), b32, 0, stream>>>(
      exo_proj, (long long)C * S, Wp_exo, 0,
      tokx, E, 1, (long long)S * E,
      nul, nul, nul, nul, 1.f, S, E, C);
  dsag_gemm<1, S, E, 1><<<dim3(13, 32, EB), b32, 0, stream>>>(
      ego_proj, (long long)C * S, Wp_ego, 0,
      tokg, E, 1, (long long)S * E,
      nul, nul, nul, nul, 1.f, S, E, C);

  // 4) pooled + normalized image features
  size_t shsz = (size_t)(E + 33) * sizeof(float);
  dsag_img_pool_norm<<<BEXO, b256, shsz, stream>>>(tokx, imgx, S, E);
  dsag_img_pool_norm<<<EB,   b256, shsz, stream>>>(tokg, imgg, S, E);

  // 5) CLIP logits: imgn @ txtn^T * exp(ls)   A:(E,1) B:(1,E)
  dsag_gemm<E, 1, 1, E><<<dim3(6, 2, 1), b32, 0, stream>>>(
      imgx, 0, txtn, 0, o_lpi_exo, NC, 1, 0,
      nul, nul, nul, ls_exo, 1.f, BEXO, NC, E);
  dsag_gemm<E, 1, 1, E><<<dim3(2, 2, 1), b32, 0, stream>>>(
      imgg, 0, txtn, 0, o_lpi, NC, 1, 0,
      nul, nul, nul, ls_ego, 1.f, EB, NC, E);
  dsag_transpose<<<(BEXO * NC + 255) / 256, b256, 0, stream>>>(o_lpi_exo, o_lpt_exo, BEXO, NC);
  dsag_transpose<<<(EB * NC + 255) / 256, b256, 0, stream>>>(o_lpi, o_lpt, EB, NC);

  // 6) attention maps
  dsag_att<<<BEXO, b256, 0, stream>>>(tokx, txtn, label, ls_exo, attx, nullptr, S, E);
  dsag_att<<<EB,   b256, 0, stream>>>(tokg, txtn, label, ls_ego, attg, csg, S, E);

  // 7) exo branch: weighted spatial mean -> affordance GEMM -> head GEMM
  dsag_wmean<<<(BEXO * C + 255) / 256, b256, 0, stream>>>(
      exo_proj, attx, wmx, (long long)BEXO * C, C, S);
  dsag_gemm<C, 1, 1, C><<<dim3(6, 16, 1), b32, 0, stream>>>(
      wmx, 0, Waff_exo, 0, affx, C2, 1, 0,
      nul, baff_exo, nul, nul, 1.f, BEXO, C2, C);
  dsag_gemm<C2, 1, 1, C2><<<dim3(6, 2, 1), b32, 0, stream>>>(
      affx, 0, fce_w, 0, o_exo_score, NC, 1, 0,
      nul, fce_b, nul, nul, 1.f, BEXO, NC, C2);

  // 8) ego attention scores -> softmax -> sinkhorn -> sym -> mu@mu
  dsag_gemm<E, 1, 1, E><<<dim3(13, 7, EB), b32, 0, stream>>>(
      tokg, (long long)S * E, tokg, (long long)S * E,
      muA, S, 1, (long long)S * S,
      nul, nul, nul, nul, 0.03125f /*1/sqrt(1024)*/, S, S, E);
  dsag_softmax_rows<<<(EB * S + 7) / 8, b256, 0, stream>>>(muA, (long long)EB * S, S);
  for (int it = 0; it < 3; ++it) {
    dsag_mu_colnorm<<<(EB * S + 255) / 256, b256, 0, stream>>>(muA, EB, S);
    dsag_mu_rownorm<<<(EB * S + 255) / 256, b256, 0, stream>>>(muA, EB, S);
  }
  dsag_mu_sym<<<((long long)EB * S * S + 255) / 256, b256, 0, stream>>>(
      muA, muB, (long long)EB * S * S, S);
  dsag_gemm<S, 1, S, 1><<<dim3(13, 7, EB), b32, 0, stream>>>(
      muB, (long long)S * S, muB, (long long)S * S,
      o_mu, S, 1, (long long)S * S,
      nul, nul, nul, nul, 1.f, S, S, S);

  // 9) ego_branch = W_aff_ego @ (colscale * ego_proj) + b (scale in epilogue)
  dsag_gemm<C, 1, S, 1><<<dim3(32, 7, EB), b32, 0, stream>>>(
      Waff_ego, 0, ego_proj, (long long)C * S,
      o_ebr, S, 1, (long long)C2 * S,
      baff_ego, nul, csg, nul, 1.f, C2, S, C);

  // 10) ego score head + cam + sim loss
  dsag_mean_last<<<(EB * C2 + 255) / 256, b256, 0, stream>>>(o_ebr, egm, (long long)EB * C2, S);
  dsag_gemm<C2, 1, 1, C2><<<dim3(2, 2, 1), b32, 0, stream>>>(
      egm, 0, fc_w, 0, o_ego_score, NC, 1, 0,
      nul, fc_b, nul, nul, 1.f, EB, NC, C2);
  dsag_cam<<<(EB * S + 255) / 256, b256, 0, stream>>>(o_ebr, fc_w, label, o_cam, EB, C2, S);
  dsag_sim_loss<<<BEXO, b32, 0, stream>>>(attg, attx, o_sim, S);
}
